// MMDLoss_71536975282442
// MI455X (gfx1250) — compile-verified, matrix operand order checked by reading, not run on hardware
//
#include <hip/hip_runtime.h>
#include <hip/hip_bf16.h>

#define N_PTS 8192
#define DIMS  128
#define BT    128                 // block tile (rows/cols of output per block)
#define NB    (N_PTS / BT)        // 64 block tiles per dimension
#define TRI   (NB * (NB + 1) / 2) // 2080 upper-triangular tiles
#define NBLK  (2 * TRI + NB * NB) // 8256 blocks: xx-tri, yy-tri, xy-full

typedef __attribute__((ext_vector_type(16))) __bf16   v16bf;
typedef __attribute__((ext_vector_type(8)))  __bf16   v8bf;
typedef __attribute__((ext_vector_type(8)))  float    v8f;
typedef __attribute__((ext_vector_type(4)))  unsigned v4u;
typedef __attribute__((ext_vector_type(8)))  int      v8i;
typedef __attribute__((ext_vector_type(4)))  int      v4i;

// ---------- fp32 <-> bf16 helpers (RNE, bit-exact) ----------
static __device__ __forceinline__ __bf16 f2bf(float f) {
    unsigned u = __float_as_uint(f);
    unsigned r = u + 0x7fffu + ((u >> 16) & 1u);   // round-to-nearest-even
    unsigned short hb = (unsigned short)(r >> 16);
    __bf16 h;
    __builtin_memcpy(&h, &hb, 2);
    return h;
}
static __device__ __forceinline__ float bf2f(__bf16 h) {
    unsigned short hb;
    __builtin_memcpy(&hb, &h, 2);
    return __uint_as_float(((unsigned)hb) << 16);
}

// ---------------------------------------------------------------------------
// Kernel 1: split fp32 rows into bf16 hi/lo, compute row squared norms.
// ---------------------------------------------------------------------------
__global__ __launch_bounds__(256) void mmd_prep_kernel(
    const float* __restrict__ x, const float* __restrict__ y,
    __bf16* __restrict__ xhi, __bf16* __restrict__ xlo,
    __bf16* __restrict__ yhi, __bf16* __restrict__ ylo,
    float* __restrict__ x2, float* __restrict__ y2)
{
    int gwave = (int)(blockIdx.x * (blockDim.x >> 5)) + (threadIdx.x >> 5);
    int lane  = threadIdx.x & 31;
    if (gwave >= 2 * N_PTS) return;
    int which = gwave >> 13;            // 0 = x, 1 = y
    int row   = gwave & (N_PTS - 1);

    const float* src = which ? y : x;
    __bf16* hi = which ? yhi : xhi;
    __bf16* lo = which ? ylo : xlo;
    float*  nr = which ? y2  : x2;

    const float* r = src + (size_t)row * DIMS;
    float s = 0.0f;
#pragma unroll
    for (int c = 0; c < DIMS / 32; ++c) {
        int col = lane + c * 32;
        float f = r[col];
        __bf16 h = f2bf(f);
        float rem = f - bf2f(h);
        __bf16 l = f2bf(rem);
        hi[(size_t)row * DIMS + col] = h;
        lo[(size_t)row * DIMS + col] = l;
        s += f * f;
    }
#pragma unroll
    for (int off = 16; off > 0; off >>= 1) s += __shfl_down(s, off, 32);
    if (lane == 0) nr[row] = s;
}

// ---------------------------------------------------------------------------
// TDM: DMA one contiguous 32KB block (4096 x 8-byte elements) global -> LDS.
// D# built per CDNA5 ISA (count=1, type=2, data_size=8B,
// tile_dim0=tensor_dim0=stride0=4096, tile_dim1=1).
// ---------------------------------------------------------------------------
#if __has_builtin(__builtin_amdgcn_tensor_load_to_lds) && __has_builtin(__builtin_amdgcn_s_wait_tensorcnt)
#define USE_TDM 1
#else
#define USE_TDM 0
#endif

#if USE_TDM
static __device__ __forceinline__ void tdm_copy_32k(const void* gsrc, void* ldst) {
    unsigned long long ga = (unsigned long long)gsrc;
    unsigned           la = (unsigned)(unsigned long long)ldst;  // addr[31:0] = LDS offset
    v4u g0;
    g0[0] = 1u;                                                // count=1 (valid user D#)
    g0[1] = la;                                                // lds_addr
    g0[2] = (unsigned)(ga & 0xffffffffu);                      // global_addr[31:0]
    g0[3] = (unsigned)((ga >> 32) & 0x1ffffffu) | (2u << 30);  // addr[56:32] | type=2
    v8i g1;
    g1[0] = (int)(3u << 16);        // workgroup_mask=0 | data_size=3 (8 bytes)
    g1[1] = (int)(4096u << 16);     // tensor_dim0[15:0]=4096 (bits 79:48 low half)
    g1[2] = (int)(1u << 16);        // tensor_dim0[31:16]=0 | tensor_dim1[15:0]=1
    g1[3] = (int)(4096u << 16);     // tensor_dim1[31:16]=0 | tile_dim0=4096
    g1[4] = 1;                      // tile_dim1=1 | tile_dim2=0
    g1[5] = 4096;                   // tensor_dim0_stride[31:0]
    g1[6] = 0;                      // stride0[47:32]=0 | stride1 low (unused)
    g1[7] = 0;
    v4i z4 = {0, 0, 0, 0};
#if defined(__clang_major__) && __clang_major__ >= 23
    v8i z8 = {0, 0, 0, 0, 0, 0, 0, 0};
    __builtin_amdgcn_tensor_load_to_lds(g0, g1, z4, z4, z8, 0);
#else
    __builtin_amdgcn_tensor_load_to_lds(g0, g1, z4, z4, 0);
#endif
}
#endif

// ---------------------------------------------------------------------------
// Fragment load from LDS: 16-bit A/B layout (ISA 7.12.2).
// ---------------------------------------------------------------------------
static __device__ __forceinline__ v16bf load_frag(const __bf16* p) {
    v8bf a = *(const v8bf*)p;          // K = kb .. kb+7
    v8bf b = *(const v8bf*)(p + 16);   // K = kb+16 .. kb+23
    v16bf r;
#pragma unroll
    for (int e = 0; e < 8; ++e) { r[e] = a[e]; r[e + 8] = b[e]; }
    return r;
}

// ---------------------------------------------------------------------------
// Kernel 2: fused Gram + RBF + partial reduction.
// Symmetry: xx/yy computed on upper-triangular blocks only (off-diag x2).
// ---------------------------------------------------------------------------
__global__ __launch_bounds__(256) void mmd_gemm_kernel(
    const __bf16* __restrict__ xhi, const __bf16* __restrict__ xlo,
    const __bf16* __restrict__ yhi, const __bf16* __restrict__ ylo,
    const float* __restrict__ x2, const float* __restrict__ y2,
    float* __restrict__ partials)
{
    extern __shared__ uint4 smem4[];
    __bf16* sAh = (__bf16*)smem4;              // 128 x 128 bf16 = 32KB each
    __bf16* sAl = sAh + BT * DIMS;
    __bf16* sBh = sAl + BT * DIMS;
    __bf16* sBl = sBh + BT * DIMS;
    __shared__ float wsum[8];

    // ---- decode block -> (which, bi, bj) -------------------------------
    int t = (int)blockIdx.x;
    int which, bi, bj;
    if (t < 2 * TRI) {
        which = (t < TRI) ? 0 : 1;
        int u = (t < TRI) ? t : (t - TRI);
        int bjj = (int)((__builtin_sqrtf(8.0f * (float)u + 1.0f) - 1.0f) * 0.5f);
        while ((bjj + 1) * (bjj + 2) / 2 <= u) ++bjj;   // exact correction
        while (bjj * (bjj + 1) / 2 > u) --bjj;
        bi = u - bjj * (bjj + 1) / 2;                   // bi <= bj
        bj = bjj;
    } else {
        which = 2;
        int u = t - 2 * TRI;
        bi = u / NB;
        bj = u % NB;
    }

    const __bf16* Ah = (which == 1) ? yhi : xhi;
    const __bf16* Al = (which == 1) ? ylo : xlo;
    const __bf16* Bh = (which == 0) ? xhi : yhi;
    const __bf16* Bl = (which == 0) ? xlo : ylo;
    const float*  a2 = (which == 1) ? y2 : x2;
    const float*  b2 = (which == 0) ? x2 : y2;

    // ---- stage A/B blocks into LDS -------------------------------------
#if USE_TDM
    if (threadIdx.x < 32) {   // one wave issues the DMAs (EXEC ignored by TDM)
        tdm_copy_32k(Ah + (size_t)bi * BT * DIMS, sAh);
        tdm_copy_32k(Al + (size_t)bi * BT * DIMS, sAl);
        tdm_copy_32k(Bh + (size_t)bj * BT * DIMS, sBh);
        tdm_copy_32k(Bl + (size_t)bj * BT * DIMS, sBl);
        __builtin_amdgcn_s_wait_tensorcnt(0);
    }
#else
    {
        const uint4* gA_h = (const uint4*)(Ah + (size_t)bi * BT * DIMS);
        const uint4* gA_l = (const uint4*)(Al + (size_t)bi * BT * DIMS);
        const uint4* gB_h = (const uint4*)(Bh + (size_t)bj * BT * DIMS);
        const uint4* gB_l = (const uint4*)(Bl + (size_t)bj * BT * DIMS);
        uint4* dA_h = (uint4*)sAh;
        uint4* dA_l = (uint4*)sAl;
        uint4* dB_h = (uint4*)sBh;
        uint4* dB_l = (uint4*)sBl;
        const int CH = BT * DIMS * 2 / 16;  // 2048 x 16B chunks per array
        for (int idx = threadIdx.x; idx < CH; idx += 256) {
            dA_h[idx] = gA_h[idx];
            dA_l[idx] = gA_l[idx];
            dB_h[idx] = gB_h[idx];
            dB_l[idx] = gB_l[idx];
        }
    }
#endif
    __syncthreads();

    int lane = threadIdx.x & 31;
    int wave = threadIdx.x >> 5;
    int wm   = wave & 3;        // 4 row groups  * 32 rows = 128
    int wn   = wave >> 2;       // 2 col groups  * 64 cols = 128
    int lrow = lane & 15;
    int lgrp = lane >> 4;

    v8f acc[2][4] = {};

#pragma unroll
    for (int ko = 0; ko < DIMS / 32; ++ko) {
        int kb = ko * 32 + lgrp * 8;
        v16bf ah[2], al[2], bh[4], bl[4];
#pragma unroll
        for (int i = 0; i < 2; ++i) {
            int rowoff = (wm * 32 + i * 16 + lrow) * DIMS + kb;
            ah[i] = load_frag(sAh + rowoff);
            al[i] = load_frag(sAl + rowoff);
        }
#pragma unroll
        for (int j = 0; j < 4; ++j) {
            int rowoff = (wn * 64 + j * 16 + lrow) * DIMS + kb;
            bh[j] = load_frag(sBh + rowoff);
            bl[j] = load_frag(sBl + rowoff);
        }
#pragma unroll
        for (int i = 0; i < 2; ++i) {
#pragma unroll
            for (int j = 0; j < 4; ++j) {
                acc[i][j] = __builtin_amdgcn_wmma_f32_16x16x32_bf16(
                    false, ah[i], false, bh[j], (short)0, acc[i][j], false, false);
                acc[i][j] = __builtin_amdgcn_wmma_f32_16x16x32_bf16(
                    false, ah[i], false, bl[j], (short)0, acc[i][j], false, false);
                acc[i][j] = __builtin_amdgcn_wmma_f32_16x16x32_bf16(
                    false, al[i], false, bh[j], (short)0, acc[i][j], false, false);
            }
        }
    }

    // ---- epilogue: exp(-(|a|^2 + |b|^2 - 2 a.b) / dim^2), local reduce --
    const float INV = 1.0f / (float)(DIMS * DIMS);   // 1/16384
    int gi0 = bi * BT + wm * 32;
    int gj0 = bj * BT + wn * 64;
    float psum = 0.0f;
#pragma unroll
    for (int j = 0; j < 4; ++j) {
        float b2v = b2[gj0 + j * 16 + lrow];
#pragma unroll
        for (int i = 0; i < 2; ++i) {
#pragma unroll
            for (int r = 0; r < 8; ++r) {
                float a2v = a2[gi0 + i * 16 + r + lgrp * 8];
                float sq  = a2v + b2v - 2.0f * acc[i][j][r];
                psum += __expf(-sq * INV);
            }
        }
    }
#pragma unroll
    for (int off = 16; off > 0; off >>= 1) psum += __shfl_down(psum, off, 32);
    if (lane == 0) wsum[wave] = psum;
    __syncthreads();
    if (threadIdx.x == 0) {
        float tot = 0.0f;
#pragma unroll
        for (int w = 0; w < 8; ++w) tot += wsum[w];  // fixed order: deterministic
        const float invN2 = 1.0f / 67108864.0f;      // 1/(8192*8192)
        float weight;
        if (which == 2)       weight = -2.0f * invN2;          // cross term
        else if (bi == bj)    weight =  invN2;                 // diagonal block
        else                  weight =  2.0f * invN2;          // symmetric pair
        partials[blockIdx.x] = tot * weight;
    }
}

// ---------------------------------------------------------------------------
// Kernel 3: deterministic final reduction of per-block partials.
// ---------------------------------------------------------------------------
__global__ __launch_bounds__(256) void mmd_reduce_kernel(
    const float* __restrict__ partials, float* __restrict__ out)
{
    __shared__ float sh[256];
    float s = 0.0f;
    for (int i = threadIdx.x; i < NBLK; i += 256) s += partials[i];
    sh[threadIdx.x] = s;
    __syncthreads();
    for (int off = 128; off > 0; off >>= 1) {
        if ((int)threadIdx.x < off) sh[threadIdx.x] += sh[threadIdx.x + off];
        __syncthreads();
    }
    if (threadIdx.x == 0) out[0] = sh[0];
}

// ---------------------------------------------------------------------------
extern "C" void kernel_launch(void* const* d_in, const int* in_sizes, int n_in,
                              void* d_out, int out_size, void* d_ws, size_t ws_size,
                              hipStream_t stream) {
    // setup_inputs order: inputs (= y), true_samples (= x)
    const float* y_in = (const float*)d_in[0];
    const float* x_in = (const float*)d_in[1];

    // workspace carve (all 16B aligned): 4 x 2MB bf16 splits, norms, partials
    const size_t MAT = (size_t)N_PTS * DIMS;          // 1,048,576 elements
    char* w = (char*)d_ws;
    __bf16* xhi = (__bf16*)(w);
    __bf16* xlo = (__bf16*)(w + MAT * 2);
    __bf16* yhi = (__bf16*)(w + MAT * 4);
    __bf16* ylo = (__bf16*)(w + MAT * 6);
    float*  x2  = (float*)(w + MAT * 8);
    float*  y2  = x2 + N_PTS;
    float*  partials = y2 + N_PTS;

    // 1) split + norms: 2*8192 rows, 8 rows (waves) per block
    mmd_prep_kernel<<<(2 * N_PTS) / 8, 256, 0, stream>>>(
        x_in, y_in, xhi, xlo, yhi, ylo, x2, y2);

    // 2) fused Gram + RBF: 8256 blocks (triangular xx/yy + full xy), 128KB LDS
    const size_t lds_bytes = 4 * (size_t)BT * DIMS * sizeof(__bf16); // 131072
    mmd_gemm_kernel<<<NBLK, 256, lds_bytes, stream>>>(
        xhi, xlo, yhi, ylo, x2, y2, partials);

    // 3) deterministic scalar reduction
    mmd_reduce_kernel<<<1, 256, 0, stream>>>(partials, (float*)d_out);
}
